// INT4MultiHeadAttention_84516366451354
// MI455X (gfx1250) — compile-verified, matrix operand order checked by reading, not run on hardware
//
#include <hip/hip_runtime.h>
#include <hip/hip_bf16.h>

// ---------------------------------------------------------------------------
// INT4-quantized multi-head attention for MI455X (gfx1250, wave32, WMMA).
// B=2, T=2048, D=1024, H=16, HD=64. All matmuls via v_wmma_f32_16x16x32_f16.
// Tile staging uses CDNA5 async global->LDS copies (ASYNCcnt) with double
// buffering so the WMMA pipe overlaps with the tile fills.
// ---------------------------------------------------------------------------

typedef __attribute__((ext_vector_type(16))) _Float16 v16h;
typedef __attribute__((ext_vector_type(4)))  _Float16 v4h;
typedef __attribute__((ext_vector_type(8)))  float    v8f;

#define WMMA_F16(a, b, c) \
    __builtin_amdgcn_wmma_f32_16x16x32_f16(false, (a), false, (b), (short)0, (c), false, false)

// --- CDNA5 async global->LDS copy (GVS mode), tracked by ASYNCcnt -----------
__device__ __forceinline__ void async_load_b128(uint32_t lds_byte_addr,
                                                const void* sbase,
                                                uint32_t byte_off) {
    asm volatile("global_load_async_to_lds_b128 %0, %1, %2"
                 :
                 : "v"(lds_byte_addr), "v"(byte_off), "s"(sbase)
                 : "memory");
}
__device__ __forceinline__ void wait_async0() {
    asm volatile("s_wait_asynccnt 0x0" ::: "memory");
}
__device__ __forceinline__ uint32_t lds_addr_of(const void* p) {
    return (uint32_t)(uintptr_t)p;  // LDS generic addr: low 32 bits = LDS offset
}

// --- WMMA fragment loaders (ISA 7.12.2 wave32 layouts) ----------------------
// A-matrix 16x32 f16: lane row = row0 + (lane&15); VGPR j holds K pair
//   kk = ((j>>2)<<4) + ((lane>=16)?8:0) + ((j&3)<<1)
__device__ __forceinline__ v16h load_frag_row(const _Float16* base, int ld,
                                              int row0, int k0, int lane) {
    const int r  = row0 + (lane & 15);
    const int hi = (lane >> 4) << 3;          // 0 or 8
    v16h f;
#pragma unroll
    for (int j = 0; j < 8; ++j) {
        const int kk = k0 + ((j >> 2) << 4) + hi + ((j & 3) << 1);
        f[2 * j]     = base[r * ld + kk];
        f[2 * j + 1] = base[r * ld + kk + 1];
    }
    return f;
}

// B-matrix (KxN) where memory holds it row-major as [k][n]: lane = column.
__device__ __forceinline__ v16h load_frag_col(const _Float16* base, int ld,
                                              int col0, int k0, int lane) {
    const int c  = col0 + (lane & 15);
    const int hi = (lane >> 4) << 3;
    v16h f;
#pragma unroll
    for (int j = 0; j < 8; ++j) {
        const int kk = k0 + ((j >> 2) << 4) + hi + ((j & 3) << 1);
        f[2 * j]     = base[kk * ld + c];
        f[2 * j + 1] = base[(kk + 1) * ld + c];
    }
    return f;
}

// --- fp32 -> f16 convert ----------------------------------------------------
__global__ __launch_bounds__(256) void cvt_f32_f16(const float* __restrict__ X,
                                                   _Float16* __restrict__ Y, int n) {
    int i = (blockIdx.x * 256 + threadIdx.x) * 4;
    if (i < n) {
        float4 v = *(const float4*)(X + i);
        v4h o = { (_Float16)v.x, (_Float16)v.y, (_Float16)v.z, (_Float16)v.w };
        *(v4h*)(Y + i) = o;
    }
}

// --- int4 quant-dequant per 64-element head vector, fp32 in -> f16 out ------
__global__ __launch_bounds__(256) void quant_int4_rows64(const float* __restrict__ X,
                                                         _Float16* __restrict__ Y,
                                                         int nrows) {
    const int row  = (int)((blockIdx.x * blockDim.x + threadIdx.x) >> 5);
    const int lane = threadIdx.x & 31;
    if (row >= nrows) return;
    const float* x = X + (size_t)row * 64;
    float a = x[lane], b = x[lane + 32];
    float am = fmaxf(fabsf(a), fabsf(b));
#pragma unroll
    for (int m = 1; m <= 16; m <<= 1) am = fmaxf(am, __shfl_xor(am, m, 32));
    am = fmaxf(am, 1e-5f);
    const float scale = am * (1.0f / 7.0f);
    const float inv   = 7.0f / am;
    float qa = fminf(fmaxf(rintf(a * inv), -8.0f), 7.0f) * scale;
    float qb = fminf(fmaxf(rintf(b * inv), -8.0f), 7.0f) * scale;
    _Float16* y = Y + (size_t)row * 64;
    y[lane]      = (_Float16)qa;
    y[lane + 32] = (_Float16)qb;
}

// --- GEMM: C(f32, MxN) = A(f16, MxK) @ W(f16, NxK)^T + bias -----------------
// Workgroup tile 128x128, 8 waves (4 along M x 2 along N), 32x64 per wave.
// Double-buffered async tile fills: load stage s^1 while WMMAs run on stage s.
#define GTM 128
#define GTN 128
#define GTK 32

__global__ __launch_bounds__(256) void gemm_f16_nt_wmma(
    const _Float16* __restrict__ A, const _Float16* __restrict__ W,
    const float* __restrict__ bias, float* __restrict__ C,
    int M, int N, int K) {
    __shared__ __attribute__((aligned(16))) _Float16 As[2][GTM][GTK];
    __shared__ __attribute__((aligned(16))) _Float16 Ws[2][GTN][GTK];

    const int tid  = threadIdx.x;
    const int lane = tid & 31;
    const int wave = tid >> 5;   // 0..7
    const int wm   = wave >> 1;  // 0..3
    const int wn   = wave & 1;   // 0..1
    const int m0   = blockIdx.y * GTM;
    const int n0   = blockIdx.x * GTN;

    v8f acc[2][4] = {};

    // async stage fill: 128 rows x 32 halves per tile = 512 b128 copies;
    // 256 threads x 2 per matrix.
    auto stage_load = [&](int s, int k0) {
#pragma unroll
        for (int i = 0; i < 2; ++i) {
            int v = tid + i * 256;
            int r = v >> 2;
            int c = (v & 3) * 8;
            async_load_b128(lds_addr_of(&As[s][r][c]), A,
                            (uint32_t)(((m0 + r) * K + k0 + c) * 2));
            async_load_b128(lds_addr_of(&Ws[s][r][c]), W,
                            (uint32_t)(((n0 + r) * K + k0 + c) * 2));
        }
    };

    stage_load(0, 0);
    int s = 0;
    for (int k0 = 0; k0 < K; k0 += GTK) {
        wait_async0();      // stage s resident
        __syncthreads();    // all waves done writing s / done reading s^1
        if (k0 + GTK < K) stage_load(s ^ 1, k0 + GTK);  // overlap with WMMAs

        v16h af[2], wf[4];
#pragma unroll
        for (int i = 0; i < 2; ++i)
            af[i] = load_frag_row(&As[s][0][0], GTK, wm * 32 + i * 16, 0, lane);
#pragma unroll
        for (int j = 0; j < 4; ++j)
            wf[j] = load_frag_row(&Ws[s][0][0], GTK, wn * 64 + j * 16, 0, lane);
#pragma unroll
        for (int i = 0; i < 2; ++i)
#pragma unroll
            for (int j = 0; j < 4; ++j)
                acc[i][j] = WMMA_F16(af[i], wf[j], acc[i][j]);
        s ^= 1;
    }

    const int cn = lane & 15;
    const int hb = (lane >> 4) << 3;  // M offset 0 or 8 within a 16x16 tile
#pragma unroll
    for (int i = 0; i < 2; ++i)
#pragma unroll
        for (int j = 0; j < 4; ++j)
#pragma unroll
            for (int r = 0; r < 8; ++r) {
                int m = m0 + wm * 32 + i * 16 + r + hb;
                int n = n0 + wn * 64 + j * 16 + cn;
                C[(size_t)m * N + n] = acc[i][j][r] + bias[n];
            }
}

// --- Flash attention over the int4-dequantized f16 KV cache -----------------
// Block = 128 threads (4 waves). Each wave owns a 16-query tile of one (b,h).
// Double-buffered async K/V chunk staging; online softmax; WMMA for QK^T & PV.
__global__ __launch_bounds__(128) void attn_int4_wmma(
    const _Float16* __restrict__ Qh, const _Float16* __restrict__ Kh,
    const _Float16* __restrict__ Vh, _Float16* __restrict__ Oh, int Tlen) {
    __shared__ __attribute__((aligned(16))) _Float16 Ks[2][32][64];
    __shared__ __attribute__((aligned(16))) _Float16 Vs[2][32][64];
    __shared__ __attribute__((aligned(16))) _Float16 Ps[4][16][32];

    const int tid  = threadIdx.x;
    const int lane = tid & 31;
    const int wave = tid >> 5;
    const int b    = blockIdx.z;
    const int h    = blockIdx.y;
    const int q0   = blockIdx.x * 64 + wave * 16;
    const int Dm   = 1024;
    const size_t rowbase = (size_t)b * Tlen * Dm;
    const int hcol = h * 64;

    // Q tile fragments: A 16x64 split into two 16x32 fragments (read once).
    v16h qf[2];
    {
        const _Float16* qb = Qh + rowbase + (size_t)q0 * Dm + hcol;
        qf[0] = load_frag_row(qb, Dm, 0, 0, lane);
        qf[1] = load_frag_row(qb, Dm, 0, 32, lane);
    }

    const _Float16* kbase = Kh + rowbase + hcol;
    const _Float16* vbase = Vh + rowbase + hcol;

    // async chunk fill: 32 keys x 64 halves = 256 b128 copies per buffer;
    // 128 threads x 2 per buffer.
    auto chunk_load = [&](int s, int kc) {
#pragma unroll
        for (int i = 0; i < 2; ++i) {
            int v  = tid + i * 128;
            int rr = v >> 3;
            int cc = (v & 7) * 8;
            uint32_t go = (uint32_t)((((kc + rr) * Dm) + cc) * 2);
            async_load_b128(lds_addr_of(&Ks[s][rr][cc]), kbase, go);
            async_load_b128(lds_addr_of(&Vs[s][rr][cc]), vbase, go);
        }
    };

    float mrow[8], lrow[8];
#pragma unroll
    for (int r = 0; r < 8; ++r) { mrow[r] = -3.0e38f; lrow[r] = 0.0f; }
    v8f acc[4] = {};
    const float sc = 0.125f;  // HD^-0.5

    chunk_load(0, 0);
    int s = 0;
    for (int kc = 0; kc < Tlen; kc += 32) {
        wait_async0();
        __syncthreads();
        if (kc + 32 < Tlen) chunk_load(s ^ 1, kc + 32);  // overlap with compute

        // scores: 16 queries x 32 keys (two 16x16 tiles), contraction over HD=64
        v8f s0 = {}, s1 = {};
        {
            v16h kf;
            kf = load_frag_row(&Ks[s][0][0], 64, 0, 0, lane);   // keys 0-15, d 0-31
            s0 = WMMA_F16(qf[0], kf, s0);
            kf = load_frag_row(&Ks[s][0][0], 64, 0, 32, lane);  // keys 0-15, d 32-63
            s0 = WMMA_F16(qf[1], kf, s0);
            kf = load_frag_row(&Ks[s][0][0], 64, 16, 0, lane);  // keys 16-31
            s1 = WMMA_F16(qf[0], kf, s1);
            kf = load_frag_row(&Ks[s][0][0], 64, 16, 32, lane);
            s1 = WMMA_F16(qf[1], kf, s1);
        }

        // online softmax; C layout: lane half holds rows (r + 8*(lane>=16))
        float alpha[8];
        const int pm = ((lane >> 4) << 3);
        const int pn = lane & 15;
#pragma unroll
        for (int r = 0; r < 8; ++r) {
            float t0 = s0[r] * sc, t1 = s1[r] * sc;
            float mx = fmaxf(t0, t1);
#pragma unroll
            for (int m = 1; m <= 8; m <<= 1) mx = fmaxf(mx, __shfl_xor(mx, m, 32));
            float mnew = fmaxf(mrow[r], mx);
            float p0 = __expf(t0 - mnew);
            float p1 = __expf(t1 - mnew);
            float ps = p0 + p1;
#pragma unroll
            for (int m = 1; m <= 8; m <<= 1) ps += __shfl_xor(ps, m, 32);
            float a  = __expf(mrow[r] - mnew);
            lrow[r]  = lrow[r] * a + ps;
            mrow[r]  = mnew;
            alpha[r] = a;
            Ps[wave][r + pm][pn]      = (_Float16)p0;
            Ps[wave][r + pm][pn + 16] = (_Float16)p1;
        }
#pragma unroll
        for (int j = 0; j < 4; ++j) {
            v8f t = acc[j];
#pragma unroll
            for (int r = 0; r < 8; ++r) t[r] *= alpha[r];
            acc[j] = t;
        }
        // P (16x32) @ V (32x64): V stored [key][d] -> column-per-lane B frags
        v16h pf = load_frag_row(&Ps[wave][0][0], 32, 0, 0, lane);
#pragma unroll
        for (int j = 0; j < 4; ++j) {
            v16h vf = load_frag_col(&Vs[s][0][0], 64, j * 16, 0, lane);
            acc[j] = WMMA_F16(pf, vf, acc[j]);
        }
        s ^= 1;
    }

    const int n  = lane & 15;
    const int hb = (lane >> 4) << 3;
#pragma unroll
    for (int j = 0; j < 4; ++j)
#pragma unroll
        for (int r = 0; r < 8; ++r) {
            float o = acc[j][r] / lrow[r];
            size_t g = rowbase + (size_t)(q0 + r + hb) * Dm + hcol + j * 16 + n;
            Oh[g] = (_Float16)o;
        }
}

// ---------------------------------------------------------------------------
extern "C" void kernel_launch(void* const* d_in, const int* in_sizes, int n_in,
                              void* d_out, int out_size, void* d_ws, size_t ws_size,
                              hipStream_t stream) {
    (void)in_sizes; (void)n_in; (void)out_size; (void)ws_size;
    const float* query = (const float*)d_in[0];
    const float* key_  = (const float*)d_in[1];
    const float* value = (const float*)d_in[2];
    const float* Wq = (const float*)d_in[3];
    const float* bq = (const float*)d_in[4];
    const float* Wk = (const float*)d_in[5];
    const float* bk = (const float*)d_in[6];
    const float* Wv = (const float*)d_in[7];
    const float* bv = (const float*)d_in[8];
    const float* Wo = (const float*)d_in[9];
    const float* bo = (const float*)d_in[10];
    float* out = (float*)d_out;

    const int Bc = 2, Tc = 2048, Dc = 1024;
    const int M = Bc * Tc;           // 4096
    const size_t MD = (size_t)M * Dc;
    const size_t DD = (size_t)Dc * Dc;

    char* ws = (char*)d_ws;
    size_t off = 0;
    auto carve = [&](size_t bytes) {
        void* p = ws + off;
        off += (bytes + 255) & ~(size_t)255;
        return p;
    };
    _Float16* xq_h = (_Float16*)carve(MD * 2);
    _Float16* xk_h = (_Float16*)carve(MD * 2);
    _Float16* xv_h = (_Float16*)carve(MD * 2);
    _Float16* Wq_h = (_Float16*)carve(DD * 2);
    _Float16* Wk_h = (_Float16*)carve(DD * 2);
    _Float16* Wv_h = (_Float16*)carve(DD * 2);
    _Float16* Wo_h = (_Float16*)carve(DD * 2);
    float*    q_f  = (float*)carve(MD * 4);
    float*    k_f  = (float*)carve(MD * 4);
    float*    v_f  = (float*)carve(MD * 4);
    _Float16* q_h  = (_Float16*)carve(MD * 2);
    _Float16* k_h  = (_Float16*)carve(MD * 2);
    _Float16* v_h  = (_Float16*)carve(MD * 2);
    _Float16* o_h  = (_Float16*)carve(MD * 2);

    // fp32 -> f16 conversions
    {
        int n = (int)MD;
        int g = (n / 4 + 255) / 256;
        cvt_f32_f16<<<g, 256, 0, stream>>>(query, xq_h, n);
        cvt_f32_f16<<<g, 256, 0, stream>>>(key_,  xk_h, n);
        cvt_f32_f16<<<g, 256, 0, stream>>>(value, xv_h, n);
        int nw = (int)DD;
        int gw = (nw / 4 + 255) / 256;
        cvt_f32_f16<<<gw, 256, 0, stream>>>(Wq, Wq_h, nw);
        cvt_f32_f16<<<gw, 256, 0, stream>>>(Wk, Wk_h, nw);
        cvt_f32_f16<<<gw, 256, 0, stream>>>(Wv, Wv_h, nw);
        cvt_f32_f16<<<gw, 256, 0, stream>>>(Wo, Wo_h, nw);
    }

    // Q/K/V projections: (4096x1024) @ (1024x1024)^T
    dim3 ggrid(Dc / GTN, M / GTM);
    gemm_f16_nt_wmma<<<ggrid, 256, 0, stream>>>(xq_h, Wq_h, bq, q_f, M, Dc, Dc);
    gemm_f16_nt_wmma<<<ggrid, 256, 0, stream>>>(xk_h, Wk_h, bk, k_f, M, Dc, Dc);
    gemm_f16_nt_wmma<<<ggrid, 256, 0, stream>>>(xv_h, Wv_h, bv, v_f, M, Dc, Dc);

    // int4 quant-dequant of K/V (per 64-elem head vector); Q just converts.
    {
        int nrows = (int)(MD / 64);                 // 65536
        int g = (nrows * 32 + 255) / 256;
        quant_int4_rows64<<<g, 256, 0, stream>>>(k_f, k_h, nrows);
        quant_int4_rows64<<<g, 256, 0, stream>>>(v_f, v_h, nrows);
        int n = (int)MD;
        cvt_f32_f16<<<(n / 4 + 255) / 256, 256, 0, stream>>>(q_f, q_h, n);
    }

    // Attention: grid (T/64, H, B), 4 waves per block.
    dim3 agrid(Tc / 64, 16, Bc);
    attn_int4_wmma<<<agrid, 128, 0, stream>>>(q_h, k_h, v_h, o_h, Tc);

    // Output projection -> fp32 d_out
    gemm_f16_nt_wmma<<<ggrid, 256, 0, stream>>>(o_h, Wo_h, bo, out, M, Dc, Dc);
}
